// BERTBiLSTM_75453985456705
// MI455X (gfx1250) — matrix operand order repaired and checked by reading, not compiled, not run on hardware
//
#include <hip/hip_runtime.h>
#include <hip/hip_bf16.h>

// ---------------------------------------------------------------------------
// MI455X (gfx1250) implementation of the BERT-BiLSTM tagging forward pass.
// wave32, WMMA f16 16x16x32 -> f32 accumulate for all GEMMs.
// GEMM uses double-buffered LDS staging; async global->LDS (ASYNCcnt) when
// the toolchain exposes the gfx1250 builtins, else a pipelined reg path.
// ---------------------------------------------------------------------------

typedef __attribute__((ext_vector_type(16))) _Float16 v16h;
typedef __attribute__((ext_vector_type(8)))  float    v8f;

#if defined(__has_builtin)
#if __has_builtin(__builtin_amdgcn_global_load_async_to_lds_b128) && \
    __has_builtin(__builtin_amdgcn_s_wait_asynccnt)
#define HAVE_ASYNC_LDS 1
#endif
#endif
#ifndef HAVE_ASYNC_LDS
#define HAVE_ASYNC_LDS 0
#endif

#if HAVE_ASYNC_LDS
// builtin expects: (v4i global*, v4i shared*, imm offset, imm cpol)
// where v4i is a gcc-style vector of 4 ints (per hipcc diagnostic).
typedef int v4i_b128 __attribute__((vector_size(16)));
typedef __attribute__((address_space(1))) v4i_b128* gas_b128;
typedef __attribute__((address_space(3))) v4i_b128* las_b128;
#define GLD_ASYNC_B128(gp, lp)                                               \
    __builtin_amdgcn_global_load_async_to_lds_b128(                          \
        (gas_b128)(unsigned long long)(gp), (las_b128)(lp), 0, 0)
#endif

struct H16 { uint4 lo, hi; };

__device__ __forceinline__ v16h make_frag(uint4 a, uint4 b) {
    H16 t; t.lo = a; t.hi = b;
    return __builtin_bit_cast(v16h, t);
}

__device__ __forceinline__ float sigf(float x) { return 1.0f / (1.0f + expf(-x)); }

// ----------------------- conversion helpers -------------------------------

__global__ void cvt_f32_to_f16(const float* __restrict__ src,
                               _Float16* __restrict__ dst, int n) {
    int i = blockIdx.x * blockDim.x + threadIdx.x;
    if (i < n) dst[i] = (_Float16)src[i];
}

// src [R][C] f32 -> dst [C][R] f16 (transpose W_gate into N-major layout)
__global__ void cvt_transpose_f16(const float* __restrict__ src,
                                  _Float16* __restrict__ dst, int R, int C) {
    int i = blockIdx.x * blockDim.x + threadIdx.x;
    if (i < R * C) {
        int r = i / C, c = i % C;
        dst[(size_t)c * R + r] = (_Float16)src[i];
    }
}

__global__ void zero_acc(float* p) { if (threadIdx.x < 4) p[threadIdx.x] = 0.0f; }

__global__ void finalize_loss(const float* __restrict__ lacc, float* __restrict__ out) {
    out[0] = lacc[0] / lacc[1] + lacc[2] / lacc[3];
}

// ----------------------- lower head: z_t, CE1, alpha, z_s' -----------------
__global__ __launch_bounds__(256)
void head_lower(const float* __restrict__ x, const int* __restrict__ amask,
                const int* __restrict__ labels_n, const float* __restrict__ Wn,
                const float* __restrict__ bn, const float* __restrict__ Wtrans,
                float* __restrict__ alpha, float* __restrict__ zsp,
                float* __restrict__ lacc) {
    int row  = blockIdx.x * 8 + (threadIdx.x >> 5);  // r = b*512 + s
    int lane = threadIdx.x & 31;
    const float* xr = x + (size_t)row * 768;
    float a0 = 0.f, a1 = 0.f, a2 = 0.f;
    for (int k = lane; k < 768; k += 32) {
        float xv = xr[k];
        a0 += xv * Wn[k * 3 + 0];
        a1 += xv * Wn[k * 3 + 1];
        a2 += xv * Wn[k * 3 + 2];
    }
    for (int off = 16; off; off >>= 1) {
        a0 += __shfl_xor(a0, off, 32);
        a1 += __shfl_xor(a1, off, 32);
        a2 += __shfl_xor(a2, off, 32);
    }
    if (lane == 0) {
        float z0 = a0 + bn[0], z1 = a1 + bn[1], z2 = a2 + bn[2];
        float mx = fmaxf(z0, fmaxf(z1, z2));
        float e0 = expf(z0 - mx), e1 = expf(z1 - mx), e2 = expf(z2 - mx);
        float se = e0 + e1 + e2;
        float lse = mx + logf(se);
        int   lbl = labels_n[row];
        float zl  = (lbl == 0) ? z0 : (lbl == 1) ? z1 : z2;
        float m   = (float)amask[row];
        atomicAdd(&lacc[0], (lse - zl) * m);
        atomicAdd(&lacc[1], m);
        float p0 = e0 / se, p1 = e1 / se, p2 = e2 / se;
        alpha[row] = 0.5f * (p0 * p0 + p1 * p1 + p2 * p2);   // EPS = 0.5
        for (int j = 0; j < 7; ++j)
            zsp[row * 7 + j] = z0 * Wtrans[j] + z1 * Wtrans[7 + j] + z2 * Wtrans[14 + j];
    }
}

// ----------------------- generic WMMA GEMM --------------------------------
// C[M,N](f32) = A[M,K](f16,row-major) x W[N,K](f16,row-major)^T (+bias)(+sigmoid)
// block tile 64x256, 8 waves, each wave 2x4 tiles of 16x16, K chunk 32,
// double-buffered LDS with async global->LDS staging.
__global__ __launch_bounds__(256)
void wmma_gemm_nt(const _Float16* __restrict__ A, const _Float16* __restrict__ W,
                  const float* __restrict__ bias, float* __restrict__ out,
                  int M, int N, int K, int remap, int act) {
    __shared__ __align__(16) _Float16 shA[2][64 * 32];
    __shared__ __align__(16) _Float16 shW[2][256 * 32];
    const int tid  = threadIdx.x;
    const int lane = tid & 31, wid = tid >> 5;
    const int wm = wid >> 2, wn = wid & 3;
    const int rowBase = blockIdx.x * 64;
    const int colBase = blockIdx.y * 256;
    const int hi = lane >> 4, l16 = lane & 15;

    // staging coordinates: one b128 of A + four b128 of W per thread per chunk
    const int sr = tid >> 2, sk = (tid & 3) * 8;  // halves
    const _Float16* gA  = A + (size_t)(rowBase + sr) * K + sk;
    const _Float16* gW0 = W + (size_t)(colBase + sr) * K + sk;
    const _Float16* gW1 = W + (size_t)(colBase + 64 + sr) * K + sk;
    const _Float16* gW2 = W + (size_t)(colBase + 128 + sr) * K + sk;
    const _Float16* gW3 = W + (size_t)(colBase + 192 + sr) * K + sk;

    v8f acc[2][4] = {};
    const int nk = K >> 5;

#if HAVE_ASYNC_LDS
    auto stage = [&](int buf, int kc) {
        GLD_ASYNC_B128(gA + kc,  &((uint4*)shA[buf])[tid]);
        GLD_ASYNC_B128(gW0 + kc, &((uint4*)shW[buf])[tid]);
        GLD_ASYNC_B128(gW1 + kc, &((uint4*)shW[buf])[256 + tid]);
        GLD_ASYNC_B128(gW2 + kc, &((uint4*)shW[buf])[512 + tid]);
        GLD_ASYNC_B128(gW3 + kc, &((uint4*)shW[buf])[768 + tid]);
    };
    stage(0, 0);
    __builtin_amdgcn_s_wait_asynccnt(0);
    __syncthreads();
#else
    uint4 ra, rw0, rw1, rw2, rw3;
    auto ldg = [&](int kc) {
        ra  = *(const uint4*)(gA + kc);
        rw0 = *(const uint4*)(gW0 + kc);
        rw1 = *(const uint4*)(gW1 + kc);
        rw2 = *(const uint4*)(gW2 + kc);
        rw3 = *(const uint4*)(gW3 + kc);
    };
    auto sts = [&](int buf) {
        ((uint4*)shA[buf])[tid]       = ra;
        ((uint4*)shW[buf])[tid]       = rw0;
        ((uint4*)shW[buf])[256 + tid] = rw1;
        ((uint4*)shW[buf])[512 + tid] = rw2;
        ((uint4*)shW[buf])[768 + tid] = rw3;
    };
    ldg(0);
    sts(0);
    __syncthreads();
#endif

    for (int ki = 0; ki < nk; ++ki) {
        const int cur = ki & 1;
        const bool has_next = (ki + 1 < nk);
#if HAVE_ASYNC_LDS
        if (has_next) stage(cur ^ 1, (ki + 1) << 5);   // overlap with WMMA below
#else
        if (has_next) ldg((ki + 1) << 5);              // loads in flight over WMMA
#endif
        v16h af[2];
        for (int mt = 0; mt < 2; ++mt) {
            const char* p = (const char*)shA[cur] +
                            (wm * 32 + mt * 16 + l16) * 64 + hi * 16;
            af[mt] = make_frag(*(const uint4*)p, *(const uint4*)(p + 32));
        }
        for (int nt = 0; nt < 4; ++nt) {
            const char* p = (const char*)shW[cur] +
                            (wn * 64 + nt * 16 + l16) * 64 + hi * 16;
            v16h wf = make_frag(*(const uint4*)p, *(const uint4*)(p + 32));
            acc[0][nt] = __builtin_amdgcn_wmma_f32_16x16x32_f16(
                false, af[0], false, wf, (short)0, acc[0][nt], false, false);
            acc[1][nt] = __builtin_amdgcn_wmma_f32_16x16x32_f16(
                false, af[1], false, wf, (short)0, acc[1][nt], false, false);
        }
        if (has_next) {
#if HAVE_ASYNC_LDS
            __builtin_amdgcn_s_wait_asynccnt(0);
#else
            sts(cur ^ 1);
#endif
            __syncthreads();
        }
    }

    for (int mt = 0; mt < 2; ++mt)
        for (int nt = 0; nt < 4; ++nt) {
            int col = colBase + wn * 64 + nt * 16 + l16;
            float bv = bias ? bias[col] : 0.0f;
            for (int e = 0; e < 8; ++e) {
                int r = rowBase + wm * 32 + mt * 16 + hi * 8 + e;
                float v = acc[mt][nt][e] + bv;
                if (act) v = sigf(v);
                int orow = remap ? ((r & 511) * 32 + (r >> 9)) : r;  // (b,s)->(s,b)
                out[(size_t)orow * N + col] = v;
            }
        }
}

// ----------------------- persistent bidirectional LSTM scan ----------------
// 2 blocks (dir), 24 waves. Wave u owns hidden units [u*16,u*16+16) for all 4
// gates and all 32 batch rows: acc layout == cell-state register layout, so
// the gate combine is shuffle-free. h (f16) lives in LDS; w_hh f16 streamed
// from L2 (2.25 MB resident in 192 MB L2).
__global__ __launch_bounds__(768)
void lstm_scan(const _Float16* __restrict__ whh_f, const _Float16* __restrict__ whh_b,
               const float* __restrict__ pre_f, const float* __restrict__ pre_b,
               float* __restrict__ hs32, _Float16* __restrict__ hs16) {
    __shared__ __align__(16) _Float16 hl[32 * 384];
    const int dir = blockIdx.x;
    const _Float16* whh = dir ? whh_b : whh_f;
    const float*    pre = dir ? pre_b : pre_f;
    const int tid = threadIdx.x, lane = tid & 31, u = tid >> 5;
    const int hi = lane >> 4, l16 = lane & 15;
    const int j = u * 16 + l16;  // hidden unit column

    for (int i = tid; i < 32 * 384; i += 768) hl[i] = (_Float16)0.0f;
    float creg[2][8];
    for (int mt = 0; mt < 2; ++mt)
        for (int e = 0; e < 8; ++e) creg[mt][e] = 0.0f;
    __syncthreads();

    for (int step = 0; step < 512; ++step) {
        const int s = dir ? (511 - step) : step;
        v8f acc[4][2] = {};
        for (int kc = 0; kc < 12; ++kc) {  // K = 384
            v16h af[2];
            for (int mt = 0; mt < 2; ++mt) {
                const char* p = (const char*)hl + (mt * 16 + l16) * 768 + kc * 64 + hi * 16;
                af[mt] = make_frag(*(const uint4*)p, *(const uint4*)(p + 32));
            }
            for (int g = 0; g < 4; ++g) {  // i,f,g,o gate blocks of w_hh
                const char* p = (const char*)whh +
                                (size_t)(g * 384 + j) * 768 + kc * 64 + hi * 16;
                v16h wf = make_frag(*(const uint4*)p, *(const uint4*)(p + 32));
                acc[g][0] = __builtin_amdgcn_wmma_f32_16x16x32_f16(
                    false, af[0], false, wf, (short)0, acc[g][0], false, false);
                acc[g][1] = __builtin_amdgcn_wmma_f32_16x16x32_f16(
                    false, af[1], false, wf, (short)0, acc[g][1], false, false);
            }
        }
        __syncthreads();  // all waves done reading hl
        for (int mt = 0; mt < 2; ++mt)
            for (int e = 0; e < 8; ++e) {
                int m = mt * 16 + hi * 8 + e;                 // batch row
                size_t base = (size_t)(s * 32 + m) * 1536;    // pre = x-proj + bias
                float iv = acc[0][mt][e] + pre[base + j];
                float fv = acc[1][mt][e] + pre[base + 384 + j];
                float gv = acc[2][mt][e] + pre[base + 768 + j];
                float ov = acc[3][mt][e] + pre[base + 1152 + j];
                float c = sigf(fv) * creg[mt][e] + sigf(iv) * tanhf(gv);
                float h = sigf(ov) * tanhf(c);
                creg[mt][e] = c;
                hl[m * 384 + j] = (_Float16)h;
                size_t ob = (size_t)(s * 32 + m) * 768 + dir * 384 + j;
                hs32[ob] = h;
                hs16[ob] = (_Float16)h;
            }
        __syncthreads();  // hl updated for next step
    }
}

// ----------------------- highway gate scan (in-place h_s -> h_tilde) -------
__global__ void gate_scan(float* __restrict__ hs, const float* __restrict__ g) {
    int t = blockIdx.x * blockDim.x + threadIdx.x;  // 32*768 chains
    int b = t / 768, f = t % 768;
    float hprev = hs[(size_t)b * 768 + f];          // s = 0 kept as-is
    for (int s = 1; s < 512; ++s) {
        size_t idx = (size_t)(s * 32 + b) * 768 + f;
        float gv = g[idx];
        float h = gv * hs[idx] + (1.0f - gv) * hprev;
        hs[idx] = h;
        hprev = h;
    }
}

// ----------------------- upper head: z_s, blend, CE2, logits ---------------
__global__ __launch_bounds__(256)
void head_upper(const float* __restrict__ ht, const float* __restrict__ Wu,
                const float* __restrict__ bu, const float* __restrict__ alpha,
                const float* __restrict__ zsp, const int* __restrict__ amask,
                const int* __restrict__ labels, float* __restrict__ out,
                float* __restrict__ lacc) {
    int row  = blockIdx.x * 8 + (threadIdx.x >> 5);  // r = b*512 + s
    int lane = threadIdx.x & 31;
    int b = row >> 9, s = row & 511;
    const float* hr = ht + (size_t)(s * 32 + b) * 768;  // time-major storage
    float acc[7] = {};
    for (int k = lane; k < 768; k += 32) {
        float hv = hr[k];
        for (int n = 0; n < 7; ++n) acc[n] += hv * Wu[k * 7 + n];
    }
    for (int off = 16; off; off >>= 1)
        for (int n = 0; n < 7; ++n) acc[n] += __shfl_xor(acc[n], off, 32);
    if (lane == 0) {
        float al = alpha[row];
        float z[7], mx = -1e30f;
        for (int n = 0; n < 7; ++n) {
            z[n] = zsp[row * 7 + n] * al + (acc[n] + bu[n]) * (1.0f - al);
            mx = fmaxf(mx, z[n]);
        }
        float se = 0.f;
        for (int n = 0; n < 7; ++n) se += expf(z[n] - mx);
        float lse = mx + logf(se);
        int   l = labels[row];
        float m = (float)amask[row];
        atomicAdd(&lacc[2], (lse - z[l]) * m);
        atomicAdd(&lacc[3], m);
        for (int n = 0; n < 7; ++n) out[1 + (size_t)row * 7 + n] = z[n];
    }
}

// ---------------------------------------------------------------------------

extern "C" void kernel_launch(void* const* d_in, const int* in_sizes, int n_in,
                              void* d_out, int out_size, void* d_ws, size_t ws_size,
                              hipStream_t stream) {
    const float* x        = (const float*)d_in[0];
    const int*   amask    = (const int*)d_in[1];
    const int*   labels   = (const int*)d_in[2];
    const int*   labels_n = (const int*)d_in[3];
    const float* Wn       = (const float*)d_in[4];
    const float* bn       = (const float*)d_in[5];
    const float* wih_f    = (const float*)d_in[6];
    const float* whh_f    = (const float*)d_in[7];
    const float* b_f      = (const float*)d_in[8];
    const float* wih_b    = (const float*)d_in[9];
    const float* whh_b    = (const float*)d_in[10];
    const float* b_b      = (const float*)d_in[11];
    const float* Wgate    = (const float*)d_in[12];
    const float* Wup      = (const float*)d_in[13];
    const float* bup      = (const float*)d_in[14];
    const float* Wtrans   = (const float*)d_in[15];
    float* out = (float*)d_out;

    const int M = 32 * 512;  // 16384 rows

    // workspace carve-up (256B aligned)
    size_t o = 0;
    auto alloc = [&](size_t bytes) -> size_t {
        size_t r = (o + 255) & ~(size_t)255; o = r + bytes; return r;
    };
    char* ws = (char*)d_ws;
    _Float16* xh      = (_Float16*)(ws + alloc((size_t)M * 768 * 2));
    _Float16* wih16f  = (_Float16*)(ws + alloc((size_t)1536 * 768 * 2));
    _Float16* wih16b  = (_Float16*)(ws + alloc((size_t)1536 * 768 * 2));
    _Float16* whh16f  = (_Float16*)(ws + alloc((size_t)1536 * 384 * 2));
    _Float16* whh16b  = (_Float16*)(ws + alloc((size_t)1536 * 384 * 2));
    _Float16* wgt16   = (_Float16*)(ws + alloc((size_t)768 * 768 * 2));
    float*    pre_f   = (float*)(ws + alloc((size_t)M * 1536 * 4));
    float*    pre_b   = (float*)(ws + alloc((size_t)M * 1536 * 4));
    float*    hs32    = (float*)(ws + alloc((size_t)M * 768 * 4));   // -> h_tilde
    _Float16* hs16    = (_Float16*)(ws + alloc((size_t)M * 768 * 2));
    float*    gbuf    = (float*)(ws + alloc((size_t)M * 768 * 4));
    float*    alpha   = (float*)(ws + alloc((size_t)M * 4));
    float*    zsp     = (float*)(ws + alloc((size_t)M * 7 * 4));
    float*    lacc    = (float*)(ws + alloc(4 * 4));

    zero_acc<<<1, 32, 0, stream>>>(lacc);

    // f16 conversions
    int nx = M * 768;
    cvt_f32_to_f16<<<(nx + 255) / 256, 256, 0, stream>>>(x, xh, nx);
    cvt_f32_to_f16<<<(1536 * 768 + 255) / 256, 256, 0, stream>>>(wih_f, wih16f, 1536 * 768);
    cvt_f32_to_f16<<<(1536 * 768 + 255) / 256, 256, 0, stream>>>(wih_b, wih16b, 1536 * 768);
    cvt_f32_to_f16<<<(1536 * 384 + 255) / 256, 256, 0, stream>>>(whh_f, whh16f, 1536 * 384);
    cvt_f32_to_f16<<<(1536 * 384 + 255) / 256, 256, 0, stream>>>(whh_b, whh16b, 1536 * 384);
    cvt_transpose_f16<<<(768 * 768 + 255) / 256, 256, 0, stream>>>(Wgate, wgt16, 768, 768);

    // lower head: z_t, CE1, alpha, z_t @ W_trans
    head_lower<<<M / 8, 256, 0, stream>>>(x, amask, labels_n, Wn, bn, Wtrans,
                                          alpha, zsp, lacc);

    // LSTM input projections (time-major output: row (b,s) -> s*32+b)
    wmma_gemm_nt<<<dim3(M / 64, 1536 / 256), 256, 0, stream>>>(
        xh, wih16f, b_f, pre_f, M, 1536, 768, /*remap=*/1, /*act=*/0);
    wmma_gemm_nt<<<dim3(M / 64, 1536 / 256), 256, 0, stream>>>(
        xh, wih16b, b_b, pre_b, M, 1536, 768, /*remap=*/1, /*act=*/0);

    // sequential bidirectional recurrence (one persistent WG per direction)
    lstm_scan<<<2, 768, 0, stream>>>(whh16f, whh16b, pre_f, pre_b, hs32, hs16);

    // highway gates: g = sigmoid(h_s @ W_gate)
    wmma_gemm_nt<<<dim3(M / 64, 768 / 256), 256, 0, stream>>>(
        hs16, wgt16, (const float*)nullptr, gbuf, M, 768, 768, /*remap=*/0, /*act=*/1);

    // gated highway scan (in place: hs32 becomes h_tilde)
    gate_scan<<<(32 * 768) / 256, 256, 0, stream>>>(hs32, gbuf);

    // upper head: z_s, alpha-blend, CE2, logits to d_out[1..]
    head_upper<<<M / 8, 256, 0, stream>>>(hs32, Wup, bup, alpha, zsp, amask,
                                          labels, out, lacc);

    finalize_loss<<<1, 1, 0, stream>>>(lacc, out);
    (void)in_sizes; (void)n_in; (void)out_size; (void)ws_size;
}
